// MinConv2dLSTMCell_58737972740760
// MI455X (gfx1250) — matrix operand order repaired and checked by reading, not compiled, not run on hardware
//
#include <hip/hip_runtime.h>
#include <hip/hip_bf16.h>

typedef __attribute__((ext_vector_type(16))) _Float16 v16h;
typedef __attribute__((ext_vector_type(8)))  float    v8f;

#define B_SZ   2
#define S_LEN  64
#define CIN    16
#define HW     64
#define HID    32
#define NIMG   (B_SZ * S_LEN)      // 128
#define OC     (4 * HID)           // 128
#define KTOT   (CIN * 9)           // 144
#define KSTEPS 5                   // ceil(144/32), K padded to 160
#define PLANE  ((size_t)NIMG * HID * HW * HW)   // 16,777,216 floats per gate plane

// stable softplus: log(1+e^y)
__device__ __forceinline__ float sp_(float y) {
    return fmaxf(y, 0.f) + __logf(1.f + __expf(-fabsf(y)));
}

// ---------------------------------------------------------------------------
// Pack conv weights into WMMA A-fragments (f16), 16x32 per (tile, kstep).
// Row permutation: tile t, row m -> hidden channel c = t*4 + (m>>2),
// gate = m&3, original oc = gate*32 + c.  This puts all 4 gates of one
// channel into consecutive rows, i.e. into ONE lane's accumulator VGPRs.
// A 16-bit 16x32 layout (ISA 7.12.2): lane l: m = l%16, khalf = l/16;
// element e (v=e/2, j=e&1): K = 16*(v/4) + 2*(v%4) + j + 8*khalf.
// ---------------------------------------------------------------------------
__global__ __launch_bounds__(256)
void mclstm_pack_weights(const float* __restrict__ W, _Float16* __restrict__ afrag) {
    int idx = blockIdx.x * 256 + threadIdx.x;            // 8 tiles * 5 ksteps * 512
    if (idx >= 8 * KSTEPS * 512) return;
    int e  = idx & 15;
    int l  = (idx >> 4) & 31;
    int ks = (idx >> 9) % KSTEPS;
    int t  = idx / (KSTEPS * 512);
    int khalf = l >> 4;
    int m     = l & 15;
    int v  = e >> 1;
    int jb = e & 1;
    int K = ks * 32 + ((v & 4) << 2) + ((v & 3) << 1) + jb + khalf * 8;
    int c    = t * 4 + (m >> 2);
    int gate = m & 3;
    int oc   = gate * HID + c;
    float val = (K < KTOT) ? W[oc * KTOT + K] : 0.f;
    afrag[idx] = (_Float16)val;
}

// ---------------------------------------------------------------------------
// Implicit-GEMM conv via WMMA + fused gate nonlinearity epilogue.
// Block = (image n, row h).  8 waves: wave wv owns oc-tile wv; 4 N-tiles
// cover the 64 pixels of row h.  B fragments (im2col, f16) staged in LDS,
// shared across all 8 waves.
// B 16-bit 32x16 layout (from ISA sparse-B analog): lane l: n = l%16,
// khalf = l/16; element e: K = 16*(e/8) + e%8 + 8*khalf (same K order as A).
// Outputs: fpP = f' , incP = i' * g(cell), soP = sigmoid(o)  (f32 planes).
// ---------------------------------------------------------------------------
__global__ __launch_bounds__(256)
void mclstm_conv_gemm_wmma(const float* __restrict__ x,
                           const float* __restrict__ bias,
                           const _Float16* __restrict__ afrag,
                           float* __restrict__ fpP,
                           float* __restrict__ incP,
                           float* __restrict__ soP) {
    __shared__ __align__(32) _Float16 bfrag[KSTEPS * 4 * 512];   // 20 KB

    const int blk = blockIdx.x;
    const int n = blk >> 6;          // image index 0..127
    const int h = blk & 63;          // output row
    const int tid = threadIdx.x;
    const float* xn = x + (size_t)n * CIN * HW * HW;

    // -------- cooperative im2col into B-fragment layout ------------------
    // idx decomposition chosen so 16 consecutive threads read 16 consecutive
    // ww for a fixed K -> coalesced 64B transactions.
    for (int idx = tid; idx < KSTEPS * 4 * 512; idx += 256) {
        int l16 = idx & 15;              // column n within tile -> pixel w
        int e   = (idx >> 4) & 15;       // element within v16h
        int lhi = (idx >> 8) & 1;        // lane group -> K half
        int j   = (idx >> 9) & 3;        // N-tile
        int ks  = idx >> 11;             // K-step
        int K = ks * 32 + lhi * 8 + (e & 7) + ((e >> 3) << 4);
        int w = j * 16 + l16;
        float v = 0.f;
        if (K < KTOT) {
            int cin = K / 9;
            int r   = K - cin * 9;
            int ky  = r / 3;
            int kx  = r - ky * 3;
            int hh = h + ky - 1;
            int ww = w + kx - 1;
            if (hh >= 0 && hh < HW && ww >= 0 && ww < HW)
                v = xn[(cin * HW + hh) * HW + ww];
        }
        bfrag[((ks * 4 + j) << 9) + ((lhi * 16 + l16) << 4) + e] = (_Float16)v;
    }
    __syncthreads();

    const int wv   = tid >> 5;       // wave id == oc tile 0..7
    const int lane = tid & 31;

    v8f acc[4];
    #pragma unroll
    for (int j = 0; j < 4; ++j) acc[j] = (v8f){};

    #pragma unroll
    for (int ks = 0; ks < KSTEPS; ++ks) {
        v16h a = *(const v16h*)(afrag + ((wv * KSTEPS + ks) << 9) + (lane << 4));
        #pragma unroll
        for (int j = 0; j < 4; ++j) {
            v16h b = *(const v16h*)(bfrag + ((ks * 4 + j) << 9) + (lane << 4));
            acc[j] = __builtin_amdgcn_wmma_f32_16x16x32_f16(
                false, a, false, b, (short)0, acc[j], false, false);
        }
    }

    // -------- fused epilogue: all 4 gates of a channel live in this lane --
    const int ncol = lane & 15;
    const int mhi  = lane >> 4;
    #pragma unroll
    for (int q = 0; q < 2; ++q) {
        int c = wv * 4 + mhi * 2 + q;            // hidden channel 0..31
        float bi = bias[0 * HID + c];
        float bf = bias[1 * HID + c];
        float bo = bias[2 * HID + c];
        float bc = bias[3 * HID + c];
        #pragma unroll
        for (int j = 0; j < 4; ++j) {
            int w = j * 16 + ncol;
            float i_g = acc[j][q * 4 + 0] + bi;
            float f_g = acc[j][q * 4 + 1] + bf;
            float o_g = acc[j][q * 4 + 2] + bo;
            float c_g = acc[j][q * 4 + 3] + bc;

            float diff = sp_(-f_g) - sp_(-i_g);
            float fp   = __expf(-sp_(diff));                       // f'
            float lg   = (c_g >= 0.f) ? __logf(c_g + 0.5f) : -sp_(-c_g);  // log g(cell)
            float inc  = __expf(lg - sp_(-diff));                  // i' * g(cell)
            float so   = 1.f / (1.f + __expf(-o_g));               // sigmoid(o)

            size_t idx = (((size_t)(n * HID + c)) << 12) + (h << 6) + w;
            fpP[idx]  = fp;
            incP[idx] = inc;
            soP[idx]  = so;
        }
    }
}

// ---------------------------------------------------------------------------
// Linear scan over S: c_s = f'_s * c_{s-1} + (i' * g)_s ; out = sigmoid(o)*c.
// f' + i' = 1 (normalized gates) and c0 = 0.5, so the linear-space recurrence
// is exactly as stable as the reference's log-space cumlogsumexp.
// One thread per (b, c, h, w); fully coalesced over hw.
// ---------------------------------------------------------------------------
__global__ __launch_bounds__(256)
void mclstm_scan(const float* __restrict__ fpP,
                 const float* __restrict__ incP,
                 const float* __restrict__ soP,
                 const float* __restrict__ c0,
                 float* __restrict__ out) {
    int gid = blockIdx.x * 256 + threadIdx.x;        // 0 .. 262143
    if (gid >= B_SZ * HID * HW * HW) return;
    int hw = gid & 4095;
    int c  = (gid >> 12) & (HID - 1);
    int b  = gid >> 17;

    float cs = c0[(((size_t)(b * HID + c)) << 12) + hw];
    float* clast = out + (size_t)B_SZ * S_LEN * HID * HW * HW;

    for (int s = 0; s < S_LEN; ++s) {
        int n = b * S_LEN + s;
        size_t idx = (((size_t)(n * HID + c)) << 12) + hw;   // == out index
        cs = fpP[idx] * cs + incP[idx];
        out[idx] = soP[idx] * cs;
    }
    clast[(((size_t)(b * HID + c)) << 12) + hw] = cs;
}

// ---------------------------------------------------------------------------
extern "C" void kernel_launch(void* const* d_in, const int* in_sizes, int n_in,
                              void* d_out, int out_size, void* d_ws, size_t ws_size,
                              hipStream_t stream) {
    const float* x    = (const float*)d_in[0];   // (2,64,16,64,64)
    const float* W    = (const float*)d_in[1];   // (128,16,3,3)
    const float* bias = (const float*)d_in[2];   // (128,)
    const float* c0   = (const float*)d_in[3];   // (2,1,32,64,64)
    float* out = (float*)d_out;                  // out (2,64,32,64,64) ++ c_last (2,1,32,64,64)

    // workspace layout: 3 f32 gate planes + f16 A-fragments
    float* fpP  = (float*)d_ws;
    float* incP = fpP + PLANE;
    float* soP  = incP + PLANE;
    _Float16* afrag = (_Float16*)(soP + PLANE);  // 3*PLANE*4 bytes offset, 32B aligned

    // 1) pack weights into WMMA A-fragments (gate-interleaved row order)
    mclstm_pack_weights<<<dim3((8 * KSTEPS * 512 + 255) / 256), dim3(256), 0, stream>>>(W, afrag);

    // 2) implicit-GEMM conv (WMMA) + fused gate nonlinearity
    mclstm_conv_gemm_wmma<<<dim3(NIMG * HW), dim3(256), 0, stream>>>(
        x, bias, afrag, fpP, incP, soP);

    // 3) sequential scan over S, write out and c_last
    mclstm_scan<<<dim3((B_SZ * HID * HW * HW) / 256), dim3(256), 0, stream>>>(
        fpP, incP, soP, c0, out);
}
